// Template_5729486373259
// MI455X (gfx1250) — matrix operand order repaired and checked by reading, not compiled, not run on hardware
//
#include <hip/hip_runtime.h>
#include <hip/hip_bf16.h>

typedef __attribute__((ext_vector_type(16))) _Float16 v16h;
typedef __attribute__((ext_vector_type(8)))  float    v8f;
typedef __attribute__((ext_vector_type(4)))  unsigned u32x4;
typedef __attribute__((ext_vector_type(8)))  unsigned u32x8;

#define BATCH    256
#define CHANNELS 512
#define HW       196                 // 14*14 spatial
#define NT       197                 // templates_b rows (n^2 + 1)
#define KP       224                 // logical K padded to 7 wmma k-steps of 32
#define KPP      232                 // padded row pitch in halves (LDS bank spread)
#define AP32     116                 // = KPP/2, row pitch in dwords
#define NP       208                 // N padded to 13 tiles of 16
#define CP       17                  // Cs row pitch (floats), conflict padding
#define TAUF     (0.5f / 196.0f)
#define CUTOFF_F 0.2f
#define BCHW     (BATCH * CHANNELS * HW)   // 25,690,112
#define BP_QW    (NP * KPP / 4)            // 12,064 8-byte elements in Bp

union V16 { v16h v; unsigned u[8]; _Float16 h[16]; };

// ---------------------------------------------------------------------------
// k0: per-template maxes, zero global-max bits, pack templates_b -> f16 with
//     the LDS-friendly pitch (KPP) baked into the global staging buffer
// ---------------------------------------------------------------------------
__global__ void k0_setup(const float* __restrict__ tf, const float* __restrict__ tb,
                         float* __restrict__ tmax, unsigned* __restrict__ gmaxBits,
                         _Float16* __restrict__ Bp) {
  int tid = threadIdx.x;
  if (tid == 0) *gmaxBits = 0u;
  for (int t = tid; t < HW; t += 256) {          // 196 templates_f rows
    const float* row = tf + t * HW;
    float m = -1e30f;
    for (int k = 0; k < HW; ++k) m = fmaxf(m, row[k]);
    tmax[t] = m;                                  // all positive (center of base)
  }
  for (int i = tid; i < NP * KPP; i += 256) {
    int n = i / KPP, k = i - n * KPP;
    float v = (n < NT && k < HW) ? tb[n * HW + k] : 0.0f;
    Bp[i] = (_Float16)v;
  }
}

// ---------------------------------------------------------------------------
// k1: one wave32 per (b,c): argmax over 196, atomicMax of selected template max
// ---------------------------------------------------------------------------
__global__ void k1_argmax(const float* __restrict__ x, int* __restrict__ idx,
                          const float* __restrict__ tmax,
                          unsigned* __restrict__ gmaxBits) {
  int gid  = blockIdx.x * blockDim.x + threadIdx.x;
  int bc   = gid >> 5;                 // wave id == (b,c) pair
  int lane = gid & 31;
  const float* row = x + (size_t)bc * HW;
  float bv = -1e30f; int bi = HW;
  for (int k = lane; k < HW; k += 32) {
    float v = row[k];
    if (v > bv) { bv = v; bi = k; }
  }
  #pragma unroll
  for (int m = 16; m > 0; m >>= 1) {
    float ov = __shfl_xor(bv, m, 32);
    int   oi = __shfl_xor(bi, m, 32);
    if (ov > bv || (ov == bv && oi < bi)) { bv = ov; bi = oi; }  // first-occurrence tie
  }
  if (lane == 0) {
    idx[bc] = bi;
    atomicMax(gmaxBits, __float_as_uint(tmax[bi]));  // uint order == float order (>=0)
  }
}

// ---------------------------------------------------------------------------
// k2: mask = relu(sel/gmax - CUTOFF)/CUTOFF ; x_masked = x * mask
// ---------------------------------------------------------------------------
__global__ void k2_mask(const float* __restrict__ x, const float* __restrict__ tf,
                        const int* __restrict__ idx,
                        const unsigned* __restrict__ gmaxBits,
                        float* __restrict__ xm, float* __restrict__ maskOut) {
  int i  = blockIdx.x * 256 + threadIdx.x;       // exact grid: BCHW/256 blocks
  int bc = i / HW;
  int hw = i - bc * HW;
  float rgmax = 1.0f / __uint_as_float(*gmaxBits);
  float sel   = tf[idx[bc] * HW + hw];
  float m     = fmaxf(sel * rgmax - CUTOFF_F, 0.0f) * (1.0f / CUTOFF_F);
  maskOut[i]  = m;
  xm[i]       = x[i] * m;
}

// ---------------------------------------------------------------------------
// k3: per-channel fused GEMM(256x197x196, f16 WMMA) + batch-softmax + MI loss.
//     B matrix staged LDS-resident by one Tensor Data Mover DMA per workgroup.
// ---------------------------------------------------------------------------
__global__ __launch_bounds__(256) void k3_loss(const float* __restrict__ xm,
                                               const _Float16* __restrict__ Bp,
                                               const float* __restrict__ pT,
                                               float* __restrict__ lossOut) {
  __shared__ unsigned            As32[256 * AP32];     // 118,784 B  A = xm*tau (f16x2)
  __shared__ __align__(16) _Float16 Bs[NP * KPP];      //  96,512 B  TDM destination
  __shared__ float               Cs[256 * CP];         //  17,408 B  one 256x16 strip
  __shared__ float               px[256];              //  p_x[b] accumulator
  __shared__ float               wS1[8];
  __shared__ float               wS2[8];

  int c    = blockIdx.x;
  int tid  = threadIdx.x;
  int lane = tid & 31;
  int w    = tid >> 5;

  // --- TDM: one tensor_load_to_lds DMA pulls the whole padded B (12,064 x 8B)
  if (tid < 32) {
    unsigned long long ga = (unsigned long long)(const void*)Bp;
    unsigned ldsOff = (unsigned)(size_t)(void*)&Bs[0];   // LDS aperture low bits
    u32x4 g0;
    g0.x = 1u;                                           // count=1 valid descriptor
    g0.y = ldsOff;                                       // lds_addr [63:32]
    g0.z = (unsigned)ga;                                 // global_addr lo
    g0.w = (unsigned)(ga >> 32) | 0x80000000u;           // global_addr hi | type=2
    u32x8 g1;
    g1[0] = 3u << 16;                                    // data_size = 8B
    g1[1] = (unsigned)BP_QW << 16;                       // tensor_dim0[15:0]
    g1[2] = 1u << 16;                                    // tensor_dim1 = 1
    g1[3] = (unsigned)BP_QW << 16;                       // tile_dim0[15:0]
    g1[4] = 0u;                                          // tile_dim1/2 unused
    g1[5] = (unsigned)BP_QW;                             // tensor_dim0_stride lo
    g1[6] = 0u;
    g1[7] = 0u;
    asm volatile("tensor_load_to_lds %0, %1" :: "s"(g0), "s"(g1) : "memory");
    __builtin_amdgcn_s_wait_tensorcnt(0);
  }

  // --- stage A into LDS (f16 pairs), wave w owns rows 32w..32w+31 -----------
  for (int r = 0; r < 32; ++r) {
    int b = w * 32 + r;
    const float* src = xm + ((size_t)b * CHANNELS + c) * HW;
    if (r + 1 < 32)
      __builtin_prefetch(xm + ((size_t)(b + 1) * CHANNELS + c) * HW, 0, 0);
    #pragma unroll
    for (int i = 0; i < 4; ++i) {
      int d = lane + 32 * i;                 // dword index within padded row
      if (d < AP32) {
        int k0 = 2 * d, k1 = 2 * d + 1;
        float v0 = (k0 < HW) ? src[k0] * TAUF : 0.0f;
        float v1 = (k1 < HW) ? src[k1] * TAUF : 0.0f;
        union { _Float16 h[2]; unsigned u; } pk;
        pk.h[0] = (_Float16)v0;
        pk.h[1] = (_Float16)v1;
        As32[b * AP32 + d] = pk.u;
      }
    }
  }
  px[tid] = 0.0f;
  __syncthreads();

  float l1 = 0.0f;   // lane0-per-wave accumulator of  sum_t p_T[t] * sum_b p log p

  for (int nt = 0; nt < 13; ++nt) {      // 13 N-tiles cover T=197 (padded 208)
    // ---------------- GEMM: wave w -> M-tiles 2w and 2w+1 -------------------
    #pragma unroll
    for (int half = 0; half < 2; ++half) {
      int b0 = (2 * w + half) * 16;
      v8f acc = {};
      int m  = lane & 15;
      int g  = (lane >> 4) * 8;                       // A k-group per ISA layout
      int kg = (lane >> 4) * 16;                      // unused; B loads use g, same k order as A
      const unsigned* arow = As32 + (b0 + m) * AP32;
      const unsigned* brow = (const unsigned*)&Bs[(nt * 16 + m) * KPP];
      #pragma unroll
      for (int kc = 0; kc < 7; ++kc) {
        int kb = kc * 32;
        V16 af, bf;
        #pragma unroll
        for (int vv = 0; vv < 4; ++vv) {
          af.u[vv]     = arow[(kb + g) / 2 + vv];          // K = kb+g   .. +7
          af.u[4 + vv] = arow[(kb + 16 + g) / 2 + vv];     // K = kb+16+g.. +7
        }
        #pragma unroll
        for (int vv = 0; vv < 4; ++vv) {
          bf.u[vv]     = brow[(kb + g) / 2 + vv];          // K = kb+g   .. +7
          bf.u[4 + vv] = brow[(kb + 16 + g) / 2 + vv];     // K = kb+16+g.. +7
        }
        acc = __builtin_amdgcn_wmma_f32_16x16x32_f16(
            false, af.v, false, bf.v, (short)0, acc, false, false);
      }
      int off = (lane >> 4) * 8;
      int jn  = lane & 15;
      #pragma unroll
      for (int r = 0; r < 8; ++r)
        Cs[(b0 + r + off) * CP + jn] = acc[r];             // VGPR r -> row r / r+8
    }
    __syncthreads();

    // ------------- softmax over b (256) per t-column; wave w -> cols 2w,2w+1
    for (int half = 0; half < 2; ++half) {
      int j = 2 * w + half;
      int t = nt * 16 + j;
      if (t < NT) {
        float v[8], e[8];
        float mx = -1e30f;
        #pragma unroll
        for (int i = 0; i < 8; ++i) {
          v[i] = Cs[(lane + 32 * i) * CP + j];
          mx = fmaxf(mx, v[i]);
        }
        #pragma unroll
        for (int s = 16; s > 0; s >>= 1) mx = fmaxf(mx, __shfl_xor(mx, s, 32));
        float se = 0.0f, pl = 0.0f;
        #pragma unroll
        for (int i = 0; i < 8; ++i) {
          e[i] = __expf(v[i] - mx);
          se += e[i];
          pl += e[i] * (v[i] - mx);
        }
        #pragma unroll
        for (int s = 16; s > 0; s >>= 1) {
          se += __shfl_xor(se, s, 32);
          pl += __shfl_xor(pl, s, 32);
        }
        float inv = 1.0f / se;
        float pt  = pT[t];
        #pragma unroll
        for (int i = 0; i < 8; ++i)                 // p_x[b] += p_T[t]*p (ds_add_f32)
          atomicAdd(&px[lane + 32 * i], pt * e[i] * inv);
        if (lane == 0)                              // sum_b p log p = pl/se - log se
          l1 += pt * (pl * inv - __logf(se));
      }
    }
    __syncthreads();   // Cs consumed; safe for next tile's GEMM stores
  }

  // ---- finalize: loss[c] = -S1 + sum_b p_x log p_x -------------------------
  if (lane == 0) wS1[w] = l1;
  float pv  = px[tid];
  float s2p = pv * __logf(pv);
  #pragma unroll
  for (int s = 16; s > 0; s >>= 1) s2p += __shfl_xor(s2p, s, 32);
  if (lane == 0) wS2[w] = s2p;
  __syncthreads();
  if (tid == 0) {
    float S1 = 0.0f, S2 = 0.0f;
    #pragma unroll
    for (int i = 0; i < 8; ++i) { S1 += wS1[i]; S2 += wS2[i]; }
    lossOut[c] = -S1 + S2;
  }
}

// ---------------------------------------------------------------------------
extern "C" void kernel_launch(void* const* d_in, const int* in_sizes, int n_in,
                              void* d_out, int out_size, void* d_ws, size_t ws_size,
                              hipStream_t stream) {
  const float* x  = (const float*)d_in[0];
  const float* tf = (const float*)d_in[1];
  const float* tb = (const float*)d_in[2];
  const float* pT = (const float*)d_in[3];

  float* xm      = (float*)d_out;                  // (B,C,14,14)
  float* maskOut = xm + (size_t)BCHW;              // (B,C,14,14)
  float* lossOut = xm + (size_t)2 * BCHW;          // (C,)

  char* ws = (char*)d_ws;
  int*      idx      = (int*)ws;                           // 131072 * 4 = 512 KB
  float*    tmax     = (float*)(ws + 524288);              // 196 floats
  unsigned* gmaxBits = (unsigned*)(ws + 524288 + 1024);    // 4 B
  _Float16* Bp       = (_Float16*)(ws + 524288 + 2048);    // 208*232*2 = 96,512 B

  k0_setup<<<1, 256, 0, stream>>>(tf, tb, tmax, gmaxBits, Bp);
  k1_argmax<<<(BATCH * CHANNELS * 32) / 256, 256, 0, stream>>>(x, idx, tmax, gmaxBits);
  k2_mask<<<BCHW / 256, 256, 0, stream>>>(x, tf, idx, gmaxBits, xm, maskOut);
  k3_loss<<<CHANNELS, 256, 0, stream>>>(xm, Bp, pT, lossOut);
}
